// SelfAttention_14474039787567
// MI455X (gfx1250) — matrix operand order repaired and checked by reading, not compiled, not run on hardware
//
#include <hip/hip_runtime.h>
#include <hip/hip_bf16.h>

// ---------------------------------------------------------------------------
// Self-attention (B=4, C=128, H=W=64 -> N=4096), flash-attention style.
//   Kernel 1: QKV projection (fp32 VALU), writes bf16 Q/K/V into d_ws.
//             Q pre-scaled by 1/N.  Q,K layout [B][N][C]; V layout [B][C][N].
//   Kernel 2: fused attention, online softmax, v_wmma_f32_16x16x32_bf16 for
//             both Q^T*K and V*P^T, double-buffered K/V tiles in LDS filled
//             by GLOBAL_LOAD_ASYNC_TO_LDS_B128 (ASYNCcnt) via inline asm.
// ---------------------------------------------------------------------------

typedef __bf16 bf16_t;
typedef bf16_t bf16x16 __attribute__((ext_vector_type(16)));
typedef float  f32x8   __attribute__((ext_vector_type(8)));

#define NPIX 4096
#define CCH  128

// 16B async copy global -> LDS, tracked by ASYNCcnt (GV addressing mode).
__device__ __forceinline__ void async_copy_b128(const void* g, void* l) {
  unsigned           lds = (unsigned)(size_t)l;           // DS byte address
  unsigned long long ga  = (unsigned long long)(size_t)g; // global address
  asm volatile("global_load_async_to_lds_b128 %0, %1, off"
               :: "v"(lds), "v"(ga) : "memory");
}
__device__ __forceinline__ void wait_asynccnt0() {
  asm volatile("s_wait_asynccnt 0" ::: "memory");
}

__device__ __forceinline__ bf16x16 pack2(uint4 a, uint4 b) {
  union { uint4 u[2]; bf16x16 v; } t;
  t.u[0] = a; t.u[1] = b;
  return t.v;
}

__device__ __forceinline__ float redmax16(float v) {
  v = fmaxf(v, __shfl_xor(v, 1, 32));
  v = fmaxf(v, __shfl_xor(v, 2, 32));
  v = fmaxf(v, __shfl_xor(v, 4, 32));
  v = fmaxf(v, __shfl_xor(v, 8, 32));
  return v;
}
__device__ __forceinline__ float redsum16(float v) {
  v += __shfl_xor(v, 1, 32);
  v += __shfl_xor(v, 2, 32);
  v += __shfl_xor(v, 4, 32);
  v += __shfl_xor(v, 8, 32);
  return v;
}

// ---------------------------------------------------------------------------
// Kernel 1: QKV projection.  grid = B * (N/64) blocks of 512 threads.
// ---------------------------------------------------------------------------
__global__ __launch_bounds__(512) void qkv_proj_kernel(
    const float* __restrict__ x,
    const float* __restrict__ wq, const float* __restrict__ bq,
    const float* __restrict__ wk, const float* __restrict__ bk,
    const float* __restrict__ wv, const float* __restrict__ bv,
    bf16_t* __restrict__ Q, bf16_t* __restrict__ K, bf16_t* __restrict__ V) {
  __shared__ __align__(16) float xs[CCH][64];

  const int b  = blockIdx.x >> 6;
  const int t0 = (blockIdx.x & 63) * 64;
  const int tid = threadIdx.x;

  #pragma unroll
  for (int kk = 0; kk < 4; ++kk) {
    int id  = tid + kk * 512;
    int c   = id >> 4;
    int off = (id & 15) << 2;
    *(float4*)&xs[c][off] =
        *(const float4*)&x[((size_t)b * CCH + c) * NPIX + t0 + off];
  }
  __syncthreads();

  const int o = tid & 127;
  const int g = tid >> 7;

  float aq[16], ak[16], av[16];
  #pragma unroll
  for (int p = 0; p < 16; ++p) { aq[p] = 0.f; ak[p] = 0.f; av[p] = 0.f; }

  const float* xg = &xs[0][g * 16];
  for (int c = 0; c < CCH; ++c) {
    float wqv = wq[(o << 7) + c];
    float wkv = wk[(o << 7) + c];
    float wvv = wv[(o << 7) + c];
    const float* xc = xg + c * 64;
    #pragma unroll
    for (int p = 0; p < 16; ++p) {
      float xv = xc[p];
      aq[p] = fmaf(wqv, xv, aq[p]);
      ak[p] = fmaf(wkv, xv, ak[p]);
      av[p] = fmaf(wvv, xv, av[p]);
    }
  }

  const float bqv = bq[o], bkv = bk[o], bvv = bv[o];
  const float invn = 1.0f / (float)NPIX;
  #pragma unroll
  for (int p = 0; p < 16; ++p) {
    int n = t0 + g * 16 + p;
    Q[((size_t)b * NPIX + n) * CCH + o] = (bf16_t)((aq[p] + bqv) * invn);
    K[((size_t)b * NPIX + n) * CCH + o] = (bf16_t)(ak[p] + bkv);
    V[((size_t)b * CCH + o) * NPIX + n] = (bf16_t)(av[p] + bvv);
  }
}

// ---------------------------------------------------------------------------
// Kernel 2: fused flash attention with double-buffered K/V tiles.
// grid = B * (N/128) blocks of 256 threads (8 waves); wave owns 16 queries.
// ---------------------------------------------------------------------------
__global__ __launch_bounds__(256) void flash_attn_kernel(
    const bf16_t* __restrict__ Q, const bf16_t* __restrict__ K,
    const bf16_t* __restrict__ V, const float* __restrict__ x,
    const float* __restrict__ gamma, float* __restrict__ out) {
  __shared__ __align__(16) bf16_t ldsK[2][32 * CCH];    // [buf][m][c]  16KB
  __shared__ __align__(16) bf16_t ldsV[2][CCH * 32];    // [buf][c][m]  16KB
  __shared__ __align__(16) bf16_t ldsP[8 * 16 * 32];    // per-wave [n][m]
  __shared__ float ldsStat[8][16];
  __shared__ float ldsL[8][16];

  const int tid  = threadIdx.x;
  const int wv   = tid >> 5;
  const int lane = tid & 31;
  const int ln   = lane & 15;
  const int hi   = lane >> 4;

  const int b     = blockIdx.x >> 5;
  const int qbase = (blockIdx.x & 31) * 128 + wv * 16;

  // per-thread tile-copy coordinates (2 K chunks + 2 V chunks of 16B)
  const int mA  = tid >> 4,          offA = (tid & 15) << 3;
  const int mB  = (tid + 256) >> 4,  offB = ((tid + 256) & 15) << 3;
  const int cA  = tid >> 2,          o2A  = (tid & 3) << 3;
  const int cB  = (tid + 256) >> 2,  o2B  = ((tid + 256) & 3) << 3;
  const bf16_t* Kb = K + (size_t)b * NPIX * CCH;
  const bf16_t* Vb = V + (size_t)b * CCH * NPIX;

  // ---- Q A-fragments ------------------------------------------------------
  const size_t qrow = ((size_t)b * NPIX + qbase + ln) * CCH;
  bf16x16 qf[4];
  #pragma unroll
  for (int kc = 0; kc < 4; ++kc) {
    const bf16_t* p = Q + qrow + kc * 32 + hi * 8;
    qf[kc] = pack2(*(const uint4*)p, *(const uint4*)(p + 16));
  }

  f32x8 O[8];
  #pragma unroll
  for (int i = 0; i < 8; ++i) O[i] = (f32x8)(0.0f);
  float mrow[8], lrow[8];
  #pragma unroll
  for (int r = 0; r < 8; ++r) { mrow[r] = -3.0e38f; lrow[r] = 0.0f; }

  bf16_t* ldsPW = &ldsP[wv * 512];

  // prologue: async-copy tile 0 into buffer 0
  async_copy_b128(Kb + (size_t)mA * CCH + offA, &ldsK[0][mA * CCH + offA]);
  async_copy_b128(Kb + (size_t)mB * CCH + offB, &ldsK[0][mB * CCH + offB]);
  async_copy_b128(Vb + (size_t)cA * NPIX + o2A, &ldsV[0][cA * 32 + o2A]);
  async_copy_b128(Vb + (size_t)cB * NPIX + o2B, &ldsV[0][cB * 32 + o2B]);
  wait_asynccnt0();
  __syncthreads();

  int cur = 0;
  for (int m0 = 0; m0 < NPIX; m0 += 32, cur ^= 1) {
    const int nxt = cur ^ 1;
    const bool have_next = (m0 + 32) < NPIX;
    const size_t knext = (size_t)(m0 + 32) * CCH;
    const size_t vnext = (size_t)(m0 + 32);

    // kick off next tile's async global->LDS copies; they complete under
    // this iteration's WMMA + softmax work.
    if (have_next) {
      async_copy_b128(Kb + knext + (size_t)mA * CCH + offA,
                      &ldsK[nxt][mA * CCH + offA]);
      async_copy_b128(Kb + knext + (size_t)mB * CCH + offB,
                      &ldsK[nxt][mB * CCH + offB]);
      async_copy_b128(Vb + (size_t)cA * NPIX + vnext + o2A,
                      &ldsV[nxt][cA * 32 + o2A]);
      async_copy_b128(Vb + (size_t)cB * NPIX + vnext + o2B,
                      &ldsV[nxt][cB * 32 + o2B]);
    }
    if (m0 + 64 < NPIX) {   // prefetch 2 tiles ahead -> global_prefetch_b8
      __builtin_prefetch(&Kb[(size_t)(m0 + 64 + (tid >> 3)) * CCH +
                             (tid & 7) * 16], 0, 1);
    }

    const bf16_t* ldsKc = &ldsK[cur][0];
    const bf16_t* ldsVc = &ldsV[cur][0];

    // ---- S = Q^T K : two 16x16 f32 tiles ----------------------------------
    f32x8 S0 = (f32x8)(0.0f), S1 = (f32x8)(0.0f);
    #pragma unroll
    for (int kc = 0; kc < 4; ++kc) {
      const bf16_t* k0 = &ldsKc[ln * CCH + kc * 32 + hi * 8];
      bf16x16 bf0 = pack2(*(const uint4*)k0, *(const uint4*)(k0 + 16));
      S0 = __builtin_amdgcn_wmma_f32_16x16x32_bf16(
          false, qf[kc], false, bf0, (short)0, S0, false, false);
      const bf16_t* k1 = &ldsKc[(16 + ln) * CCH + kc * 32 + hi * 8];
      bf16x16 bf1 = pack2(*(const uint4*)k1, *(const uint4*)(k1 + 16));
      S1 = __builtin_amdgcn_wmma_f32_16x16x32_bf16(
          false, qf[kc], false, bf1, (short)0, S1, false, false);
    }

    // ---- online softmax ---------------------------------------------------
    float alpha[8];
    #pragma unroll
    for (int r = 0; r < 8; ++r) {
      float rm = redmax16(fmaxf(S0[r], S1[r]));
      float mn = fmaxf(mrow[r], rm);
      alpha[r] = __expf(mrow[r] - mn);
      mrow[r]  = mn;
      S0[r] = __expf(S0[r] - mn);
      S1[r] = __expf(S1[r] - mn);
      lrow[r] = alpha[r] * lrow[r] + redsum16(S0[r] + S1[r]);
    }

    if (lane == 0) {
      #pragma unroll
      for (int r = 0; r < 8; ++r) ldsStat[wv][r] = alpha[r];
    }
    if (lane == 16) {
      #pragma unroll
      for (int r = 0; r < 8; ++r) ldsStat[wv][8 + r] = alpha[r];
    }
    asm volatile("s_wait_dscnt 0" ::: "memory");
    float aL = ldsStat[wv][ln];
    #pragma unroll
    for (int i = 0; i < 8; ++i)
      #pragma unroll
      for (int e = 0; e < 8; ++e) O[i][e] *= aL;

    // ---- transpose P through LDS (D-layout -> B-layout) -------------------
    #pragma unroll
    for (int r = 0; r < 8; ++r) {
      int n = r + 8 * hi;
      ldsPW[n * 32 + ln]      = (bf16_t)S0[r];
      ldsPW[n * 32 + 16 + ln] = (bf16_t)S1[r];
    }
    asm volatile("s_wait_dscnt 0" ::: "memory");
    const bf16_t* pp = &ldsPW[ln * 32 + hi * 8];
    bf16x16 pfrag = pack2(*(const uint4*)pp, *(const uint4*)(pp + 16));

    // ---- O += V * P^T -----------------------------------------------------
    #pragma unroll
    for (int ct = 0; ct < 8; ++ct) {
      const bf16_t* vp = &ldsVc[(ct * 16 + ln) * 32 + hi * 8];
      bf16x16 vfrag = pack2(*(const uint4*)vp, *(const uint4*)(vp + 16));
      O[ct] = __builtin_amdgcn_wmma_f32_16x16x32_bf16(
          false, vfrag, false, pfrag, (short)0, O[ct], false, false);
    }

    // ---- rotate buffers: own async copies done, then block-wide barrier ---
    wait_asynccnt0();
    __syncthreads();
  }

  // ---- epilogue: out = gamma * O / l + x ----------------------------------
  if (lane == 0) {
    #pragma unroll
    for (int r = 0; r < 8; ++r) ldsL[wv][r] = lrow[r];
  }
  if (lane == 16) {
    #pragma unroll
    for (int r = 0; r < 8; ++r) ldsL[wv][8 + r] = lrow[r];
  }
  asm volatile("s_wait_dscnt 0" ::: "memory");
  const float invl = 1.0f / ldsL[wv][ln];
  const float g = gamma[0];
  #pragma unroll
  for (int ct = 0; ct < 8; ++ct) {
    #pragma unroll
    for (int e = 0; e < 8; ++e) {
      int c = ct * 16 + e + 8 * hi;
      size_t idx = ((size_t)b * CCH + c) * NPIX + qbase + ln;
      out[idx] = g * O[ct][e] * invl + x[idx];
    }
  }
}

// ---------------------------------------------------------------------------
extern "C" void kernel_launch(void* const* d_in, const int* in_sizes, int n_in,
                              void* d_out, int out_size, void* d_ws,
                              size_t ws_size, hipStream_t stream) {
  const float* x     = (const float*)d_in[0];
  const float* wq    = (const float*)d_in[1];
  const float* bq    = (const float*)d_in[2];
  const float* wk    = (const float*)d_in[3];
  const float* bk    = (const float*)d_in[4];
  const float* wv    = (const float*)d_in[5];
  const float* bv    = (const float*)d_in[6];
  const float* gamma = (const float*)d_in[7];
  float* out = (float*)d_out;

  const size_t elems = (size_t)4 * NPIX * CCH;
  bf16_t* Q = (bf16_t*)d_ws;
  bf16_t* K = Q + elems;
  bf16_t* V = K + elems;

  qkv_proj_kernel<<<4 * 64, 512, 0, stream>>>(x, wq, bq, wk, bk, wv, bv,
                                              Q, K, V);
  flash_attn_kernel<<<4 * 32, 256, 0, stream>>>(Q, K, V, x, gamma, out);
}
